// GraphNetwork_4947802325662
// MI455X (gfx1250) — matrix operand order repaired and checked by reading, not compiled, not run on hardware
//
#include <hip/hip_runtime.h>

#define NEG_SLOPE 0.2f

typedef __attribute__((ext_vector_type(16))) _Float16 v16h;
typedef __attribute__((ext_vector_type(8)))  float    v8f;

__device__ __forceinline__ float lrelu(float v) { return v > 0.0f ? v : NEG_SLOPE * v; }

// ---------------- utility kernels ----------------

__global__ void k_set1(float* __restrict__ p, int n) {
  int i = blockIdx.x * blockDim.x + threadIdx.x;
  if (i < n) p[i] = 1.0f;
}

__global__ void k_recip(float* __restrict__ p, int n) {
  int i = blockIdx.x * blockDim.x + threadIdx.x;
  if (i < n) p[i] = 1.0f / p[i];
}

__global__ void k_copy(const float* __restrict__ src, float* __restrict__ dst, long long n) {
  long long i = (long long)blockIdx.x * blockDim.x + threadIdx.x;
  if (i < n) dst[i] = src[i];
}

// ---------------- degree ----------------

__global__ void k_deg(const int* __restrict__ col, float* __restrict__ deg, int e) {
  int i = blockIdx.x * blockDim.x + threadIdx.x;
  if (i < e) atomicAdd(&deg[col[i]], 1.0f);
}

// ---------------- edge scatter (self loops folded into acc init) ----------------

__global__ void k_scatter1(const int* __restrict__ row, const int* __restrict__ col,
                           const float* __restrict__ h, float* __restrict__ acc, int e) {
  int i = blockIdx.x * blockDim.x + threadIdx.x;
  if (i < e) atomicAdd(&acc[col[i]], h[row[i]]);
}

__global__ void k_scatter3(const int* __restrict__ row, const int* __restrict__ col,
                           const float* __restrict__ h, float* __restrict__ acc, int e) {
  int i = blockIdx.x * blockDim.x + threadIdx.x;
  if (i >= e) return;
  long long r = row[i], c = col[i];
  atomicAdd(&acc[c * 3 + 0], h[r * 3 + 0]);
  atomicAdd(&acc[c * 3 + 1], h[r * 3 + 1]);
  atomicAdd(&acc[c * 3 + 2], h[r * 3 + 2]);
}

// one thread per (edge, channel): a node row is one 64B cacheline, L2-resident (64MB < 192MB L2)
__global__ void k_scatter16(const int* __restrict__ row, const int* __restrict__ col,
                            const float* __restrict__ h, float* __restrict__ acc,
                            long long total) {
  long long t = (long long)blockIdx.x * blockDim.x + threadIdx.x;
  if (t >= total) return;
  int c = (int)(t & 15);
  long long e = t >> 4;
  long long r = row[e], cl = col[e];
  atomicAdd(&acc[cl * 16 + c], h[r * 16 + c]);
}

// ---------------- dense transforms (all WMMA) ----------------
//
// Shared WMMA mapping (V_WMMA_F32_16X16X32_F16, wave32):
//  A (16x32 f16): lanes 0-15 carry K={0..7} in elems 0..7, K={16..23} in elems 8..15
//                 lanes16-31 carry K={8..15} / K={24..31}
//  B (32x16 f16): column = lane&15; lanes 0-15 carry K=0..7 / K=16..23 (same order as A),
//                 lanes 16-31 carry K=8..15 / K=24..31
//  D (16x16 f32): lanes 0-15: col=lane, M=VGPR idx; lanes 16-31: col=lane-16, M=VGPR+8
// K packs [agg features | h features] so one WMMA computes agg@Wout + h@Wroot.
// Each kernel optionally duplicates its result into the next layer's accumulator
// (self-loop init), eliminating separate copy passes.

// fi = 16 -> fo = 1: full K=32 utilized; only D column 0 is meaningful.
__global__ void k_xform16_wmma(const float* __restrict__ h, const float* __restrict__ acc,
                               const float* __restrict__ invdeg,
                               const float* __restrict__ Wo, const float* __restrict__ bo,
                               const float* __restrict__ Wr,
                               float* __restrict__ out, float* __restrict__ acc_next, int n) {
  const int lane = threadIdx.x & 31;
  const int wave = threadIdx.x >> 5;
  const long long tile = (long long)blockIdx.x * (blockDim.x >> 5) + wave;
  const long long base = tile * 16;
  if (base >= n) return;                       // wave-uniform exit

  const int m    = lane & 15;
  const int half = lane >> 4;
  long long node = base + m;
  if (node >= n) node = n - 1;                 // clamp: garbage rows never stored
  const int f0 = half * 8;

  const float idg = invdeg[node];
  const float4* ap = (const float4*)(acc + node * 16 + f0);
  const float4* hp = (const float4*)(h   + node * 16 + f0);
  float4 a0 = ap[0], a1 = ap[1];
  float4 h0 = hp[0], h1 = hp[1];

  v16h A;
  A[0]  = (_Float16)(a0.x * idg); A[1]  = (_Float16)(a0.y * idg);
  A[2]  = (_Float16)(a0.z * idg); A[3]  = (_Float16)(a0.w * idg);
  A[4]  = (_Float16)(a1.x * idg); A[5]  = (_Float16)(a1.y * idg);
  A[6]  = (_Float16)(a1.z * idg); A[7]  = (_Float16)(a1.w * idg);
  A[8]  = (_Float16)h0.x; A[9]  = (_Float16)h0.y;
  A[10] = (_Float16)h0.z; A[11] = (_Float16)h0.w;
  A[12] = (_Float16)h1.x; A[13] = (_Float16)h1.y;
  A[14] = (_Float16)h1.z; A[15] = (_Float16)h1.w;

  // B: unconditional broadcast loads, branchless zero for columns != 0
  const float4* wo = (const float4*)(Wo + half * 8);
  const float4* wr = (const float4*)(Wr + half * 8);
  float4 w0 = wo[0], w1 = wo[1], w2 = wr[0], w3 = wr[1];
  const float sel = (m == 0) ? 1.0f : 0.0f;
  v16h Bm;
  Bm[0]  = (_Float16)(w0.x * sel); Bm[1]  = (_Float16)(w0.y * sel);
  Bm[2]  = (_Float16)(w0.z * sel); Bm[3]  = (_Float16)(w0.w * sel);
  Bm[4]  = (_Float16)(w1.x * sel); Bm[5]  = (_Float16)(w1.y * sel);
  Bm[6]  = (_Float16)(w1.z * sel); Bm[7]  = (_Float16)(w1.w * sel);
  Bm[8]  = (_Float16)(w2.x * sel); Bm[9]  = (_Float16)(w2.y * sel);
  Bm[10] = (_Float16)(w2.z * sel); Bm[11] = (_Float16)(w2.w * sel);
  Bm[12] = (_Float16)(w3.x * sel); Bm[13] = (_Float16)(w3.y * sel);
  Bm[14] = (_Float16)(w3.z * sel); Bm[15] = (_Float16)(w3.w * sel);

  v8f c = {};
  c = __builtin_amdgcn_wmma_f32_16x16x32_f16(false, A, false, Bm, (short)0, c, false, false);

  // D column 0: lane 0 has M=0..7, lane 16 has M=8..15
  if (m == 0) {
    const float bb = bo[0];
    float r[8];
#pragma unroll
    for (int j = 0; j < 8; ++j) { float v = c[j] + bb; r[j] = lrelu(v); }
    const long long ob = base + half * 8;
    if (ob + 8 <= n) {
      float4 s0, s1;
      s0.x = r[0]; s0.y = r[1]; s0.z = r[2]; s0.w = r[3];
      s1.x = r[4]; s1.y = r[5]; s1.z = r[6]; s1.w = r[7];
      ((float4*)(out + ob))[0] = s0;
      ((float4*)(out + ob))[1] = s1;
      if (acc_next) {                      // fused self-loop init for next layer
        ((float4*)(acc_next + ob))[0] = s0;
        ((float4*)(acc_next + ob))[1] = s1;
      }
    } else {
#pragma unroll
      for (int j = 0; j < 8; ++j)
        if (ob + j < n) {
          out[ob + j] = r[j];
          if (acc_next) acc_next[ob + j] = r[j];
        }
    }
  }
}

// fi in {1,3} -> fo = 16: K padded with zeros (agg at K=0..FI-1, h at K=16..16+FI-1).
// D gives all 16 output channels for 16 nodes; stores are fully coalesced.
template <int FI>
__global__ void k_xform_small_wmma(const float* __restrict__ h, const float* __restrict__ acc,
                                   const float* __restrict__ invdeg,
                                   const float* __restrict__ Wo, const float* __restrict__ bo,
                                   const float* __restrict__ Wr,
                                   float* __restrict__ out, float* __restrict__ acc_next, int n) {
  const int lane = threadIdx.x & 31;
  const int wave = threadIdx.x >> 5;
  const long long tile = (long long)blockIdx.x * (blockDim.x >> 5) + wave;
  const long long base = tile * 16;
  if (base >= n) return;                       // wave-uniform exit

  const int m    = lane & 15;
  const int half = lane >> 4;
  long long node = base + m;
  if (node >= n) node = n - 1;                 // clamp for loads only

  v16h A;
#pragma unroll
  for (int t = 0; t < 16; ++t) A[t] = (_Float16)0.0f;
  // only lanes 0-15 carry K=0..7 / 16..23 (FI <= 3 < 8), lanes 16-31 stay zero
  const float hsel = (half == 0) ? 1.0f : 0.0f;
  const float idg = invdeg[node] * hsel;
#pragma unroll
  for (int k = 0; k < FI; ++k) {
    A[k]     = (_Float16)(acc[node * FI + k] * idg);
    A[8 + k] = (_Float16)(h[node * FI + k] * hsel);
  }

  // B column m: elems 0..FI-1 carry K=0..FI-1 -> Wo rows; elems 8..8+FI-1 carry K=16.. -> Wr rows
  const float* wmat = half ? Wr : Wo;
  v16h Bm;
#pragma unroll
  for (int t = 0; t < 16; ++t) Bm[t] = (_Float16)0.0f;
#pragma unroll
  for (int k = 0; k < FI; ++k) { Bm[k] = (_Float16)Wo[k * 16 + m]; Bm[8 + k] = (_Float16)Wr[k * 16 + m]; }

  v8f c = {};
  c = __builtin_amdgcn_wmma_f32_16x16x32_f16(false, A, false, Bm, (short)0, c, false, false);

  // lane holds output channel m for nodes M = half*8 + j
  const float bb = bo[m];
  const long long nb = base + half * 8;
  if (nb + 8 <= n) {
#pragma unroll
    for (int j = 0; j < 8; ++j) {
      float v = lrelu(c[j] + bb);
      out[(nb + j) * 16 + m] = v;
      if (acc_next) acc_next[(nb + j) * 16 + m] = v;   // fused self-loop init
    }
  } else {
#pragma unroll
    for (int j = 0; j < 8; ++j) {
      if (nb + j < n) {
        float v = lrelu(c[j] + bb);
        out[(nb + j) * 16 + m] = v;
        if (acc_next) acc_next[(nb + j) * 16 + m] = v;
      }
    }
  }
}

// ---------------- host orchestration ----------------

extern "C" void kernel_launch(void* const* d_in, const int* in_sizes, int n_in,
                              void* d_out, int out_size, void* d_ws, size_t ws_size,
                              hipStream_t stream) {
  const float* x  = (const float*)d_in[0];
  const int*   ei = (const int*)d_in[1];
  const int N = in_sizes[0] / 3;
  const int E = in_sizes[1] / 2;
  const int* row = ei;
  const int* col = ei + E;

  auto Wp = [&](int i) { return (const float*)d_in[2 + 3 * i]; };
  auto Bp = [&](int i) { return (const float*)d_in[3 + 3 * i]; };
  auto Rp = [&](int i) { return (const float*)d_in[4 + 3 * i]; };

  // workspace: invdeg[N] | hA[16N] | acc16[16N] | accS[3N]  (~144 MB)
  if (ws_size < (size_t)N * (1 + 16 + 16 + 3) * sizeof(float)) return;
  float* invdeg = (float*)d_ws;
  float* hA     = invdeg + N;
  float* acc16  = hA + (size_t)16 * N;
  float* accS   = acc16 + (size_t)16 * N;

  float* outp = (float*)d_out;
  const int B = 256;
  auto gb = [&](long long n) { return (unsigned)((n + B - 1) / B); };

  const int tiles = (N + 15) / 16;
  const int wpb   = B / 32;                       // 8 waves / block (wave32)
  const unsigned gW = (unsigned)((tiles + wpb - 1) / wpb);

  // normalization: deg includes self loop (init 1), then invert in place
  k_set1 <<<gb(N), B, 0, stream>>>(invdeg, N);
  k_deg  <<<gb(E), B, 0, stream>>>(col, invdeg, E);
  k_recip<<<gb(N), B, 0, stream>>>(invdeg, N);

  // layer 0: fi=3 -> fo=16 (acc16 initialized by fused store for layer 1)
  k_copy    <<<gb(3LL * N), B, 0, stream>>>(x, accS, 3LL * N);
  k_scatter3<<<gb(E), B, 0, stream>>>(row, col, x, accS, E);
  k_xform_small_wmma<3><<<gW, B, 0, stream>>>(x, accS, invdeg, Wp(0), Bp(0), Rp(0), hA, acc16, N);

  for (int o = 0; o < 4; ++o) {
    // odd layer 2o+1: fi=16 -> fo=1, result goes straight into its output segment;
    // fused duplicate store seeds accS for the following even layer.
    int li = 2 * o + 1;
    float* seg = outp + (size_t)o * N;
    k_scatter16   <<<gb(16LL * E), B, 0, stream>>>(row, col, hA, acc16, 16LL * E);
    k_xform16_wmma<<<gW, B, 0, stream>>>(hA, acc16, invdeg, Wp(li), Bp(li), Rp(li),
                                         seg, (o < 3) ? accS : nullptr, N);

    if (o < 3) {
      // even layer 2o+2: fi=1 -> fo=16; fused duplicate store seeds acc16 for next odd layer
      int le = 2 * o + 2;
      k_scatter1<<<gb(E), B, 0, stream>>>(row, col, seg, accS, E);
      k_xform_small_wmma<1><<<gW, B, 0, stream>>>(seg, accS, invdeg, Wp(le), Bp(le), Rp(le),
                                                  hA, acc16, N);
    }
  }
}
